// VectorQuantizerEMA1D_38654705664775
// MI455X (gfx1250) — matrix operand order repaired and checked by reading, not compiled, not run on hardware
//
#include <hip/hip_runtime.h>
#include <hip/hip_bf16.h>
#include <stdint.h>

typedef __attribute__((ext_vector_type(16))) __bf16 v16bf;
typedef __attribute__((ext_vector_type(8)))  float  v8f;

#define VQ_B 16
#define VQ_D 64
#define VQ_T 8192
#define VQ_K 1024
#define VQ_N (VQ_B * VQ_T)      // 131072 vectors
#define VQ_EPS 1e-5f

// workspace layout (bytes)
#define WS_EH    0u              // K*D bf16  (131072 B)
#define WS_EL    131072u         // K*D bf16  (131072 B)
#define WS_ENORM 262144u         // K  f32    (4096 B)
#define WS_CNT   266240u         // K  f32    (4096 B)   -- contiguous with DW for zeroing
#define WS_DW    270336u         // K*D f32   (262144 B)
#define WS_IDX   532480u         // N  i32    (524288 B)

__device__ __forceinline__ unsigned short f32_to_bf16_rne(float f) {
  unsigned u = __float_as_uint(f);
  u += 0x7FFFu + ((u >> 16) & 1u);
  return (unsigned short)(u >> 16);
}
__device__ __forceinline__ float bf16u_to_f32(unsigned short h) {
  return __uint_as_float(((unsigned)h) << 16);
}

// ---------------- prep: bf16 hi/lo codebook + ||e_k||^2 ----------------
__global__ void vq_prep(const float* __restrict__ emb,
                        unsigned short* __restrict__ eh,
                        unsigned short* __restrict__ el,
                        float* __restrict__ enorm) {
  __shared__ float sq[VQ_D];
  const int k = blockIdx.x;
  const int d = threadIdx.x;
  float v = emb[k * VQ_D + d];
  unsigned short hi = f32_to_bf16_rne(v);
  eh[k * VQ_D + d] = hi;
  el[k * VQ_D + d] = f32_to_bf16_rne(v - bf16u_to_f32(hi));
  sq[d] = v * v;
  __syncthreads();
  if (d == 0) {
    float s = 0.f;
    for (int i = 0; i < VQ_D; ++i) s += sq[i];
    enorm[k] = s;
  }
}

__global__ void vq_zero(float* __restrict__ p, int n) {
  int i = blockIdx.x * blockDim.x + threadIdx.x;
  if (i < n) p[i] = 0.f;
}

// ---------------- assign: WMMA distance argmin ----------------
// One wave handles 32 rows (two 16-row M-tiles) of flat_x against all K codes.
// Each B-fragment load (4 KB/iter) now feeds 12 WMMAs -> 2x arithmetic
// intensity vs L0, and cA/cB give two independent WMMA dependency chains.
__global__ __launch_bounds__(256) void vq_assign(
    const float* __restrict__ x,
    const unsigned short* __restrict__ eh,
    const unsigned short* __restrict__ el,
    const float* __restrict__ enorm,
    int* __restrict__ idx) {
  const int lane    = threadIdx.x & 31;
  const int wave    = blockIdx.x * 8 + (threadIdx.x >> 5);
  const int rowbase = wave * 32;            // flat row (n = b*T + t)
  const int b  = rowbase / VQ_T;
  const int t0 = rowbase % VQ_T;            // T%32==0 -> tile stays in one b

  const int m    = lane & 15;               // A layout: lane -> M
  const int half = lane >> 4;

  union Frag { unsigned short u[16]; v16bf v; };
  Frag ah[2][2], al[2][2];                  // [subtile][K-chunk], hi/lo

  // 16-bit A 16x32 layout: elem e -> K = 16*half + e (matches B's contiguous order)
  #pragma unroll
  for (int s = 0; s < 2; ++s) {
    const int t = t0 + s * 16 + m;
    #pragma unroll
    for (int e = 0; e < 16; ++e) {
      const int kd = 16 * half + e;
      #pragma unroll
      for (int ch = 0; ch < 2; ++ch) {
        float xv = x[(size_t)(b * VQ_D + ch * 32 + kd) * VQ_T + t];
        unsigned short hi = f32_to_bf16_rne(xv);
        ah[s][ch].u[e] = hi;
        al[s][ch].u[e] = f32_to_bf16_rne(xv - bf16u_to_f32(hi));
      }
    }
  }

  float best[16];
  int   bidx[16];
  #pragma unroll
  for (int i = 0; i < 16; ++i) { best[i] = 3.4e38f; bidx[i] = 0; }

  const int n = lane & 15;                  // B/C layout: lane -> N (code within tile)
  for (int kc = 0; kc < VQ_K; kc += 16) {
    // B 32x16 layout: lane holds 16 contiguous K values at offset 16*half -> one 32B load
    const unsigned short* hb = eh + (size_t)(kc + n) * VQ_D + 16 * half;
    const unsigned short* lb = el + (size_t)(kc + n) * VQ_D + 16 * half;
    v16bf bh0 = *(const v16bf*)(hb);
    v16bf bh1 = *(const v16bf*)(hb + 32);
    v16bf bl0 = *(const v16bf*)(lb);
    v16bf bl1 = *(const v16bf*)(lb + 32);

    v8f cA = {}, cB = {};
    // interleave the two independent accumulation chains
    cA = __builtin_amdgcn_wmma_f32_16x16x32_bf16(false, al[0][0].v, false, bh0, (short)0, cA, false, false);
    cB = __builtin_amdgcn_wmma_f32_16x16x32_bf16(false, al[1][0].v, false, bh0, (short)0, cB, false, false);
    cA = __builtin_amdgcn_wmma_f32_16x16x32_bf16(false, ah[0][0].v, false, bl0, (short)0, cA, false, false);
    cB = __builtin_amdgcn_wmma_f32_16x16x32_bf16(false, ah[1][0].v, false, bl0, (short)0, cB, false, false);
    cA = __builtin_amdgcn_wmma_f32_16x16x32_bf16(false, ah[0][0].v, false, bh0, (short)0, cA, false, false);
    cB = __builtin_amdgcn_wmma_f32_16x16x32_bf16(false, ah[1][0].v, false, bh0, (short)0, cB, false, false);
    cA = __builtin_amdgcn_wmma_f32_16x16x32_bf16(false, al[0][1].v, false, bh1, (short)0, cA, false, false);
    cB = __builtin_amdgcn_wmma_f32_16x16x32_bf16(false, al[1][1].v, false, bh1, (short)0, cB, false, false);
    cA = __builtin_amdgcn_wmma_f32_16x16x32_bf16(false, ah[0][1].v, false, bl1, (short)0, cA, false, false);
    cB = __builtin_amdgcn_wmma_f32_16x16x32_bf16(false, ah[1][1].v, false, bl1, (short)0, cB, false, false);
    cA = __builtin_amdgcn_wmma_f32_16x16x32_bf16(false, ah[0][1].v, false, bh1, (short)0, cA, false, false);
    cB = __builtin_amdgcn_wmma_f32_16x16x32_bf16(false, ah[1][1].v, false, bh1, (short)0, cB, false, false);

    const float en   = enorm[kc + n];
    const int   cand = kc + n;
    union { float f[8]; v8f v; } ca, cb; ca.v = cA; cb.v = cB;
    #pragma unroll
    for (int i = 0; i < 8; ++i) {           // C layout: VGPR i -> M = i + 8*half
      float sA = en - 2.0f * ca.f[i];
      if (sA < best[i]) { best[i] = sA; bidx[i] = cand; }
      float sB = en - 2.0f * cb.f[i];
      if (sB < best[8 + i]) { best[8 + i] = sB; bidx[8 + i] = cand; }
    }
  }

  // reduce across the 16 lanes of each half (width-16 xor shuffles)
  #pragma unroll
  for (int i = 0; i < 16; ++i) {
    #pragma unroll
    for (int off = 8; off > 0; off >>= 1) {
      float ob = __shfl_xor(best[i], off, 16);
      int   oi = __shfl_xor(bidx[i], off, 16);
      if (ob < best[i] || (ob == best[i] && oi < bidx[i])) { best[i] = ob; bidx[i] = oi; }
    }
  }
  if ((lane & 15) == 0) {
    #pragma unroll
    for (int i = 0; i < 8; ++i) {
      idx[rowbase + half * 8 + i]      = bidx[i];        // subtile 0: M = i + 8*half
      idx[rowbase + 16 + half * 8 + i] = bidx[8 + i];    // subtile 1
    }
  }
}

// ---------------- scatter: quantized output + dw/counts accumulation ----------------
__global__ __launch_bounds__(256) void vq_scatter(
    const float* __restrict__ x,
    const float* __restrict__ emb,
    const int* __restrict__ idx,
    float* __restrict__ out,
    float* __restrict__ dw,
    float* __restrict__ counts) {
  const int blk = blockIdx.x;               // [0, B*D*(T/256))
  const int tc  = blk & 31;                 // T/256 = 32 chunks
  const int bd  = blk >> 5;                 // b*64 + d
  const int d   = bd & 63;
  const int b   = bd >> 6;
  const int t   = tc * 256 + threadIdx.x;
  const int k   = idx[b * VQ_T + t];
  const size_t off = (size_t)bd * VQ_T + t; // [B,D,T] layout
  float xv = x[off];
  out[off] = emb[k * VQ_D + d];             // quantized (straight-through == gather)
  atomicAdd(&dw[k * VQ_D + d], xv);
  if (d == 0) atomicAdd(&counts[k], 1.0f);
}

// ---------------- finalize: new_embeddings = dw / cluster_size ----------------
__global__ void vq_final(const float* __restrict__ dw,
                         const float* __restrict__ counts,
                         float* __restrict__ outE) {
  int i = blockIdx.x * blockDim.x + threadIdx.x;
  if (i >= VQ_K * VQ_D) return;
  int k = i >> 6;
  const float ntot = (float)VQ_N;           // sum(counts) is always N
  float cluster = (counts[k] + VQ_EPS) / (ntot + (float)VQ_K * VQ_EPS) * ntot;
  outE[i] = dw[i] / cluster;
}

extern "C" void kernel_launch(void* const* d_in, const int* in_sizes, int n_in,
                              void* d_out, int out_size, void* d_ws, size_t ws_size,
                              hipStream_t stream) {
  const float* x   = (const float*)d_in[0];   // [B, D, T] f32
  const float* emb = (const float*)d_in[1];   // [K, D]    f32
  float* out = (float*)d_out;                 // [B,D,T] ++ [K,D]
  char*  ws  = (char*)d_ws;

  unsigned short* eh     = (unsigned short*)(ws + WS_EH);
  unsigned short* el     = (unsigned short*)(ws + WS_EL);
  float*          enorm  = (float*)(ws + WS_ENORM);
  float*          counts = (float*)(ws + WS_CNT);
  float*          dw     = (float*)(ws + WS_DW);
  int*            idx    = (int*)(ws + WS_IDX);

  const int zn = VQ_K + VQ_K * VQ_D;          // counts + dw are contiguous
  vq_zero<<<(zn + 255) / 256, 256, 0, stream>>>(counts, zn);
  vq_prep<<<VQ_K, VQ_D, 0, stream>>>(emb, eh, el, enorm);
  vq_assign<<<(VQ_N / 32) / 8, 256, 0, stream>>>(x, eh, el, enorm, idx);
  vq_scatter<<<VQ_B * VQ_D * (VQ_T / 256), 256, 0, stream>>>(x, emb, idx, out, dw, counts);
  vq_final<<<(VQ_K * VQ_D + 255) / 256, 256, 0, stream>>>(
      dw, counts, out + (size_t)VQ_B * VQ_D * VQ_T);
}
